// LocalizationRoiLayer_45973329936540
// MI455X (gfx1250) — matrix operand order repaired and checked by reading, not compiled, not run on hardware
//
#include <hip/hip_runtime.h>
#include <hip/hip_bf16.h>

// ---------- types for WMMA ----------
typedef __attribute__((ext_vector_type(16))) __bf16          v16bf;
typedef __attribute__((ext_vector_type(8)))  float           v8f;
typedef __attribute__((ext_vector_type(8)))  unsigned short  v8u;
typedef __attribute__((ext_vector_type(16))) unsigned short  v16u;

// ---------- constants ----------
#define HW      64            // feature map H == W
#define CIN     512
#define CMID    256
#define KANCH   12
#define NPROP   (HW*HW*KANCH) // 49152
#define NMSMAX  300
#define OUTHW   7
#define ROIOUT  (NMSMAX*CIN*OUTHW*OUTHW) // 7526400

// workspace byte offsets (all 256-aligned)
#define OFF_FEATHWC   0u          // bf16 [64][64][512]      4,194,304 B
#define OFF_WSW       4194304u    // bf16 swizzled conv W    2,359,296 B
#define OFF_BASEHWC   6553600u    // bf16 [64][64][256]      2,097,152 B
#define OFF_W2        8650752u    // f32  [256][64]             65,536 B
#define OFF_RAW       8716288u    // f32  [4096][64]         1,048,576 B
#define OFF_BOXC      9764864u    // f32  [49152][4]           786,432 B
#define OFF_XYXY     10551296u    // f32  [49152][4]           786,432 B
#define OFF_SCORES   11337728u    // f32  [49152]              196,608 B
#define OFF_PICKS    11534336u    // int  [300]
#define OFF_ZPAD     11538432u    // bf16 [512] zeros (OOB A reads)

__device__ __forceinline__ unsigned short f2bf(float f) {
  unsigned u = __builtin_bit_cast(unsigned, f);
  unsigned r = u + 0x7FFFu + ((u >> 16) & 1u);
  return (unsigned short)(r >> 16);
}
__device__ __forceinline__ float bf2f(unsigned short b) {
  unsigned u = ((unsigned)b) << 16;
  return __builtin_bit_cast(float, u);
}

// A fragment: halves at +0 and +16 elements (K 0-7 / 16-23 or 8-15 / 24-31)
__device__ __forceinline__ v16bf ldafrag(const unsigned short* p) {
  v8u lo = *(const v8u*)p;
  v8u hi = *(const v8u*)(p + 16);
  return __builtin_bit_cast(v16bf, __builtin_shufflevector(lo, hi,
      0,1,2,3,4,5,6,7,8,9,10,11,12,13,14,15));
}
// B fragment: 16 contiguous bf16 per lane (pre-swizzled)
__device__ __forceinline__ v16bf ldbfrag(const unsigned short* p) {
  v8u lo = *(const v8u*)p;
  v8u hi = *(const v8u*)(p + 8);
  return __builtin_bit_cast(v16bf, __builtin_shufflevector(lo, hi,
      0,1,2,3,4,5,6,7,8,9,10,11,12,13,14,15));
}

// ---------- kernel 0: zero-pad region ----------
__global__ __launch_bounds__(256)
void k_zpad(unsigned int* __restrict__ zp) {
  zp[threadIdx.x] = 0u;   // 256 dwords = 512 bf16
}

// ---------- kernel 1: features fp32 CHW -> bf16 HWC ----------
__global__ __launch_bounds__(256)
void k_feat_hwc(const float* __restrict__ in, unsigned short* __restrict__ out) {
  int id = blockIdx.x * 256 + threadIdx.x;          // 2,097,152 total
  int pix = id & (HW*HW - 1);
  int c   = id >> 12;
  out[pix * CIN + c] = f2bf(in[c * (HW*HW) + pix]);
}

// ---------- kernel 1b: swizzle rpn_w (O,C,3,3) into B-fragment order ----------
// w_sw[((kc*16 + nt)*32 + lane)*16 + e] = B[kc*32 + klocal][nt*16 + (lane&15)]
// klocal = (e&7) + (lane>=16 ? 8 : 0) + (e>=8 ? 16 : 0); k = (r*3+s)*512 + c
__global__ __launch_bounds__(256)
void k_wsw(const float* __restrict__ w, unsigned short* __restrict__ w_sw) {
  int id = blockIdx.x * 256 + threadIdx.x;          // 1,179,648 total
  int e    = id & 15;
  int lane = (id >> 4) & 31;
  int nt   = (id >> 9) & 15;
  int kc   = id >> 13;
  if (kc >= 144) return;
  int klocal = (e & 7) + ((lane >> 4) << 3) + ((e >> 3) << 4);
  int rs = kc >> 4;                 // 512/32 = 16 chunks per (r,s)
  int c  = ((kc & 15) << 5) + klocal;
  int r  = rs / 3, s = rs % 3;
  int o  = nt * 16 + (lane & 15);
  float v = w[((o * CIN + c) * 3 + r) * 3 + s];
  w_sw[id] = f2bf(v);
}

// ---------- kernel 1c: pack 1x1 head weights into f32 [256][64] ----------
__global__ __launch_bounds__(256)
void k_w2(const float* __restrict__ box_w, const float* __restrict__ score_w,
          float* __restrict__ w2) {
  int id = blockIdx.x * 256 + threadIdx.x;          // 16384 total
  int n = id & 63, c = id >> 6;
  float v = 0.f;
  if (n < 48)      v = box_w[n * CMID + c];
  else if (n < 60) v = score_w[(n - 48) * CMID + c];
  w2[id] = v;
}

// ---------- kernel 2: 3x3 conv via WMMA bf16 + sched_barrier-enforced pipeline ----------
// GEMM: M=4096 (pixels), N=256, K=4608. Grid (64 rows, 2 channel halves).
// 8 waves: msub = wave&3 (16 px), nhalf = wave>>2 -> 4 N-tiles per wave.
// __builtin_amdgcn_sched_barrier(0) fences stop the machine scheduler from
// sinking the chunk-(cc+1) loads down to their use, forcing
//   L(0), L(1), W(0), L(2), W(1), ...
// so waitcnt insertion emits partial waits and loads overlap the WMMA groups.
__global__ __launch_bounds__(256, 1)
void k_conv3x3_wmma(const unsigned short* __restrict__ feat_hwc,
                    const unsigned short* __restrict__ w_sw,
                    const unsigned short* __restrict__ zpad,
                    const float* __restrict__ rpn_b,
                    unsigned short* __restrict__ base_hwc) {
  const int h     = blockIdx.x;         // output row 0..63
  const int cq    = blockIdx.y;         // channel half 0..1
  const int wave  = threadIdx.x >> 5;
  const int lane  = threadIdx.x & 31;
  const int msub  = wave & 3;
  const int nhalf = wave >> 2;
  const int mBase = msub << 4;
  const int nt0   = (cq << 3) + (nhalf << 2);   // first of 4 N-tiles
  const int lrow  = lane & 15;
  const int hi    = lane >> 4;
  const int px    = mBase + lrow;

  v8f acc0 = {}; v8f acc1 = {}; v8f acc2 = {}; v8f acc3 = {};

  for (int rs = 0; rs < 9; ++rs) {
    const int r = rs / 3, s = rs % 3;
    const int y = h + r - 1;
    const int x = px + s - 1;
    const bool ok = ((unsigned)y < (unsigned)HW) && ((unsigned)x < (unsigned)HW);
    // OOB lanes read zeros from the pad region: no divergent loads, EXEC stays full
    const unsigned short* ap = ok
        ? (feat_hwc + ((y * HW + x) * CIN + hi * 8))
        : (zpad + hi * 8);
    const unsigned short* bp = w_sw + (((rs * 16) * 16 + nt0) * 32 + lane) * 16;
    __builtin_prefetch(bp + 16 * 8192, 0, 1);   // next (r,s) slab of B

    v16bf a[2], b0[2], b1[2], b2[2], b3[2];
    // preload chunk 0 into slot 0
    a[0]  = ldafrag(ap);
    b0[0] = ldbfrag(bp);
    b1[0] = ldbfrag(bp + 512);
    b2[0] = ldbfrag(bp + 1024);
    b3[0] = ldbfrag(bp + 1536);
    #pragma unroll
    for (int cc = 0; cc < 16; ++cc) {
      const int cur = cc & 1;
      const int nxt = cur ^ 1;
      if (cc + 1 < 16) {
        const unsigned short* ap2 = ap + (cc + 1) * 32;
        const unsigned short* bp2 = bp + (cc + 1) * 8192;
        a[nxt]  = ldafrag(ap2);
        b0[nxt] = ldbfrag(bp2);
        b1[nxt] = ldbfrag(bp2 + 512);
        b2[nxt] = ldbfrag(bp2 + 1024);
        b3[nxt] = ldbfrag(bp2 + 1536);
      }
      __builtin_amdgcn_sched_barrier(0);   // loads(cc+1) stay above WMMAs(cc)
      acc0 = __builtin_amdgcn_wmma_f32_16x16x32_bf16(false, a[cur], false, b0[cur],
                                                     (short)0, acc0, false, false);
      acc1 = __builtin_amdgcn_wmma_f32_16x16x32_bf16(false, a[cur], false, b1[cur],
                                                     (short)0, acc1, false, false);
      acc2 = __builtin_amdgcn_wmma_f32_16x16x32_bf16(false, a[cur], false, b2[cur],
                                                     (short)0, acc2, false, false);
      acc3 = __builtin_amdgcn_wmma_f32_16x16x32_bf16(false, a[cur], false, b3[cur],
                                                     (short)0, acc3, false, false);
      __builtin_amdgcn_sched_barrier(0);   // WMMAs(cc) stay above loads(cc+2)
    }
  }

  // ---- epilogue: bias + ReLU, store bf16 HWC ----
  v8f accs[4] = {acc0, acc1, acc2, acc3};
  #pragma unroll
  for (int t = 0; t < 4; ++t) {
    int och = (nt0 + t) * 16 + lrow;
    float bias = rpn_b[och];
    #pragma unroll
    for (int v = 0; v < 8; ++v) {
      int prow = mBase + v + hi * 8;
      float val = accs[t][v] + bias;
      val = val > 0.f ? val : 0.f;
      base_hwc[(h * HW + prow) * CMID + och] = f2bf(val);
    }
  }
}

// ---------- kernel 3: 1x1 heads (box 48 + score 12), LDS-staged ----------
__global__ __launch_bounds__(64)
void k_heads(const unsigned short* __restrict__ base_hwc,
             const float* __restrict__ w2,
             const float* __restrict__ box_b, const float* __restrict__ score_b,
             float* __restrict__ raw) {
  __shared__ float sb[CMID];
  int m = blockIdx.x;                   // pixel 0..4095
  int tid = threadIdx.x;
  for (int i = tid; i < CMID; i += 64) sb[i] = bf2f(base_hwc[m * CMID + i]);
  __syncthreads();
  if (tid < 60) {
    float acc = (tid < 48) ? box_b[tid] : score_b[tid - 48];
    for (int c = 0; c < CMID; ++c) acc += sb[c] * w2[c * 64 + tid];
    raw[m * 64 + tid] = acc;
  }
}

// ---------- kernel 4: anchors + box transform + clip ----------
__global__ __launch_bounds__(256)
void k_proposals(const float* __restrict__ raw,
                 float* __restrict__ boxc, float* __restrict__ xyxy,
                 float* __restrict__ scores) {
  static const float AW[KANCH] = {45,90,64,90,180,128,181,362,256,362,724,512};
  static const float AH[KANCH] = {90,45,64,180,90,128,362,181,256,724,362,512};
  int p = blockIdx.x * 256 + threadIdx.x;   // 49152 total
  int m = p / KANCH, a = p % KANCH;
  int hy = m >> 6, wx = m & 63;
  float xa = 7.5f + 16.0f * (float)wx;
  float ya = 7.5f + 16.0f * (float)hy;
  float wa = AW[a], ha = AH[a];
  const float* rm = raw + m * 64;
  float tx = rm[a*4+0], ty = rm[a*4+1], tw = rm[a*4+2], th = rm[a*4+3];
  float sc = rm[48 + a];
  float xc = xa + tx * wa;
  float yc = ya + ty * ha;
  float bw = wa * expf(tw);
  float bh = ha * expf(th);
  float x1 = fminf(fmaxf(xc - bw*0.5f, 0.f), 1023.f);
  float x2 = fminf(fmaxf(xc + bw*0.5f, 0.f), 1023.f);
  float y1 = fminf(fmaxf(yc - bh*0.5f, 0.f), 1023.f);
  float y2 = fminf(fmaxf(yc + bh*0.5f, 0.f), 1023.f);
  float cw = x2 - x1, ch = y2 - y1;
  bool valid = (cw > 0.f) && (ch > 0.f);
  boxc[p*4+0] = (x1 + x2) * 0.5f;
  boxc[p*4+1] = (y1 + y2) * 0.5f;
  boxc[p*4+2] = cw;
  boxc[p*4+3] = ch;
  xyxy[p*4+0] = x1; xyxy[p*4+1] = y1; xyxy[p*4+2] = x2; xyxy[p*4+3] = y2;
  scores[p] = valid ? sc : -__builtin_inff();
}

// ---------- kernel 5: sequential NMS, single workgroup ----------
__global__ __launch_bounds__(1024)
void k_nms(const float* __restrict__ xyxy, float* __restrict__ scores,
           int* __restrict__ picks) {
  __shared__ float smax[1024];
  __shared__ int   sidx[1024];
  int tid = threadIdx.x;
  for (int it = 0; it < NMSMAX; ++it) {
    float best = -__builtin_inff();
    int bidx = 0x7fffffff;
    for (int p = tid; p < NPROP; p += 1024) {
      float s = scores[p];
      if (s > best || (s == best && p < bidx)) { best = s; bidx = p; }
    }
    smax[tid] = best; sidx[tid] = bidx;
    __syncthreads();
    for (int off = 512; off > 0; off >>= 1) {
      if (tid < off) {
        if (smax[tid+off] > smax[tid] ||
            (smax[tid+off] == smax[tid] && sidx[tid+off] < sidx[tid])) {
          smax[tid] = smax[tid+off]; sidx[tid] = sidx[tid+off];
        }
      }
      __syncthreads();
    }
    int b = sidx[0];
    if (tid == 0) picks[it] = b;
    float bx1 = xyxy[b*4+0], by1 = xyxy[b*4+1];
    float bx2 = xyxy[b*4+2], by2 = xyxy[b*4+3];
    float barea = (bx2 - bx1) * (by2 - by1);
    for (int p = tid; p < NPROP; p += 1024) {
      float x1 = xyxy[p*4+0], y1 = xyxy[p*4+1];
      float x2 = xyxy[p*4+2], y2 = xyxy[p*4+3];
      float area = (x2 - x1) * (y2 - y1);
      float ix1 = fmaxf(x1, bx1), iy1 = fmaxf(y1, by1);
      float ix2 = fminf(x2, bx2), iy2 = fminf(y2, by2);
      float inter = fmaxf(ix2 - ix1, 0.f) * fmaxf(iy2 - iy1, 0.f);
      float iou = inter / (area + barea - inter);
      if (iou > 0.7f) scores[p] = -__builtin_inff();
    }
    if (tid == 0) scores[b] = -__builtin_inff();
    __syncthreads();
  }
}

// ---------- kernel 6: ROI pool (bilinear) from original fp32 features ----------
__global__ __launch_bounds__(256)
void k_roipool(const float* __restrict__ feat, const float* __restrict__ boxc,
               const int* __restrict__ picks, float* __restrict__ out) {
  int idx = blockIdx.x * 256 + threadIdx.x;        // 7,526,400 total
  int b   = idx / (CIN * OUTHW * OUTHW);
  int rem = idx % (CIN * OUTHW * OUTHW);
  int c   = rem / (OUTHW * OUTHW);
  int bin = rem % (OUTHW * OUTHW);
  int i = bin / OUTHW, j = bin % OUTHW;
  int pk = picks[b];
  float xc = boxc[pk*4+0], yc = boxc[pk*4+1];
  float bw = boxc[pk*4+2], bh = boxc[pk*4+3];
  const float SXY = (float)HW / 1024.0f;           // 1/16
  float jj = ((float)j + 0.5f) / (float)OUTHW;
  float ii = ((float)i + 0.5f) / (float)OUTHW;
  float xs = (xc - bw*0.5f + jj*bw) * SXY - 0.5f;
  float ys = (yc - bh*0.5f + ii*bh) * SXY - 0.5f;
  float xq = fminf(fmaxf(xs, 0.f), 63.f);
  float yq = fminf(fmaxf(ys, 0.f), 63.f);
  int x0 = (int)floorf(xq), y0 = (int)floorf(yq);
  int x1 = min(x0 + 1, 63), y1 = min(y0 + 1, 63);
  float fx = xq - (float)x0, fy = yq - (float)y0;
  const float* fc = feat + c * (HW*HW);
  float f00 = fc[y0*HW + x0], f01 = fc[y0*HW + x1];
  float f10 = fc[y1*HW + x0], f11 = fc[y1*HW + x1];
  out[idx] = f00*(1.f-fy)*(1.f-fx) + f01*(1.f-fy)*fx
           + f10*fy*(1.f-fx)       + f11*fy*fx;
}

// ---------- kernel 7: emit picked boxes (center format) ----------
__global__ __launch_bounds__(256)
void k_boxout(const float* __restrict__ boxc, const int* __restrict__ picks,
              float* __restrict__ out) {
  int t = blockIdx.x * 256 + threadIdx.x;
  if (t >= NMSMAX * 4) return;
  out[ROIOUT + t] = boxc[picks[t >> 2] * 4 + (t & 3)];
}

extern "C" void kernel_launch(void* const* d_in, const int* in_sizes, int n_in,
                              void* d_out, int out_size, void* d_ws, size_t ws_size,
                              hipStream_t stream) {
  (void)in_sizes; (void)n_in; (void)out_size; (void)ws_size;
  const float* features = (const float*)d_in[0];
  const float* rpn_w    = (const float*)d_in[1];
  const float* rpn_b    = (const float*)d_in[2];
  const float* box_w    = (const float*)d_in[3];
  const float* box_b    = (const float*)d_in[4];
  const float* score_w  = (const float*)d_in[5];
  const float* score_b  = (const float*)d_in[6];
  float* out = (float*)d_out;

  char* ws = (char*)d_ws;
  unsigned short* feat_hwc = (unsigned short*)(ws + OFF_FEATHWC);
  unsigned short* w_sw     = (unsigned short*)(ws + OFF_WSW);
  unsigned short* base_hwc = (unsigned short*)(ws + OFF_BASEHWC);
  float* w2     = (float*)(ws + OFF_W2);
  float* raw    = (float*)(ws + OFF_RAW);
  float* boxc   = (float*)(ws + OFF_BOXC);
  float* xyxy   = (float*)(ws + OFF_XYXY);
  float* scores = (float*)(ws + OFF_SCORES);
  int*   picks  = (int*)(ws + OFF_PICKS);
  unsigned short* zpad = (unsigned short*)(ws + OFF_ZPAD);

  k_zpad<<<1, 256, 0, stream>>>((unsigned int*)zpad);
  k_feat_hwc<<<8192, 256, 0, stream>>>(features, feat_hwc);
  k_wsw<<<4608, 256, 0, stream>>>(rpn_w, w_sw);
  k_w2<<<64, 256, 0, stream>>>(box_w, score_w, w2);
  k_conv3x3_wmma<<<dim3(64, 2), 256, 0, stream>>>(feat_hwc, w_sw, zpad, rpn_b, base_hwc);
  k_heads<<<4096, 64, 0, stream>>>(base_hwc, w2, box_b, score_b, raw);
  k_proposals<<<192, 256, 0, stream>>>(raw, boxc, xyxy, scores);
  k_nms<<<1, 1024, 0, stream>>>(xyxy, scores, picks);
  k_roipool<<<29400, 256, 0, stream>>>(features, boxc, picks, out);
  k_boxout<<<5, 256, 0, stream>>>(boxc, picks, out);
}
